// QCNNGen130Model_65481071395304
// MI455X (gfx1250) — matrix operand order repaired and checked by reading, not compiled, not run on hardware
//
#include <hip/hip_runtime.h>
#include <hip/hip_bf16.h>

// CDNA5 / gfx1250, wave32. All matmuls via v_wmma_f32_16x16x32_f16.
// Activations kept transposed: D = W * X^T, features on VGPR index (M),
// batch rows on lanes (N). seq_len==1 => softmax==1 => attention out == V,
// so q_w/k_w are mathematically dead and skipped.
//
// Branchless repack: A-operand (weights) is zero at all K>=16 halves except
// the K=16 bias slot of in_d==16 layers, so hi-lane B content is a don't-care
// (multiplied by 0; activations are bounded so no Inf -> no 0*Inf NaN).

typedef __attribute__((ext_vector_type(16))) _Float16 v16h;
typedef __attribute__((ext_vector_type(2)))  __fp16   pk2h;   // cvt_pkrtz result type
typedef __attribute__((ext_vector_type(8)))  float    v8f;
typedef __attribute__((ext_vector_type(8)))  unsigned int v8u;

struct KParams {
  const float *in;
  const float *fm_w, *fm_b, *c1_w, *c1_b, *p1_w, *p1_b, *c2_w, *c2_b;
  const float *p2_w, *p2_b, *c3_w, *c3_b, *v_w, *v_b, *o_w, *o_b;
  const float *ln1_w, *ln1_b, *ln2_w, *ln2_b;
  const float *f1_w, *f1_b, *f2_w, *f2_b, *fs_w, *fs_b, *hd_w, *hd_b;
  float *out;
  int n_rows;
};

static __device__ __forceinline__ v8f wmma16(v16h a, v16h b) {
  v8f c = {};
  // (neg_a, A, neg_b, B, c_mod, C, reuse_a, reuse_b)
  return __builtin_amdgcn_wmma_f32_16x16x32_f16(false, a, false, b,
                                                (short)0, c, false, false);
}

// Build A-operand: augmented weight [out_d x (in_d+1)], bias in column k==in_d.
// A f16 16x32 order used here (contiguous k): lane L<16 holds row M=L, halves 0..15 = K0..15;
// lane L+16 holds row M=L, halves 0..15 = K16..31.
static __device__ __forceinline__ v16h make_wA(const float* w, const float* b,
                                               int out_d, int in_d, int lane) {
  v16h a = {};
  const int m = lane & 15;
  if (m < out_d) {
    const int kb = (lane < 16) ? 0 : 16;
#pragma unroll
    for (int j = 0; j < 16; ++j) {
      const int k = kb + j;
      if (k < in_d)        a[j] = (_Float16)w[m * in_d + k];
      else if (k == in_d)  a[j] = (_Float16)b[m];
    }
  }
  return a;
}

// Per-lane LN weight vector: element g corresponds to feature g (+8 on hi lanes).
static __device__ __forceinline__ v8f make_lnvec(const float* p, int n, int lane) {
  const int fb = (lane < 16) ? 0 : 8;
  v8f r = {};
#pragma unroll
  for (int g = 0; g < 8; ++g) { const int f = fb + g; if (f < n) r[g] = p[f]; }
  return r;
}

static __device__ __forceinline__ unsigned pkh(float a, float b) {
  pk2h t = __builtin_amdgcn_cvt_pkrtz(a, b);   // v_cvt_pk_rtz_f16_f32
  return __builtin_bit_cast(unsigned, t);
}

// Branchless repack of activation tile D (features x rows) into next B-operand.
// B f16 32x16 order used here (contiguous k, as A): lane n<16 = batch row n, halves 0..15 = K0..15;
// lanes 16..31 = K16..31 (don't-care except the augmented 1 when nf==16).
// nf = live feature count of this activation == in_d of the consuming layer.
static __device__ __forceinline__ v16h pack_act(v8f d, int nf, int lane) {
  v8u u = {};
  u[0] = pkh(d[0], d[1]);
  if (nf > 2) u[1] = pkh(d[2], d[3]);
  if (nf > 4) u[2] = pkh(d[4], d[5]);
  if (nf > 6) u[3] = pkh(d[6], d[7]);
  if (nf > 8) {  // features 8..15 live on hi lanes -> 4 dword shuffles
    u[4] = (unsigned)__shfl_xor((int)u[0], 16, 32);
    u[5] = (unsigned)__shfl_xor((int)u[1], 16, 32);
    u[6] = (unsigned)__shfl_xor((int)u[2], 16, 32);
    u[7] = (unsigned)__shfl_xor((int)u[3], 16, 32);
  }
  v16h b = __builtin_bit_cast(v16h, u);
  if (nf < 16) {
    b[nf] = (_Float16)1.0f;                  // hi-lane slot is don't-care
  } else {
    b[0] = (lane < 16) ? b[0] : (_Float16)1.0f;  // K=16 augmented 1 (cndmask)
  }
  return b;
}

// tanh on the first n elements (rest are identically zero for out_d<=4 layers).
static __device__ __forceinline__ v8f tanhN(v8f x, int n) {
  v8f r = {};
#pragma unroll
  for (int i = 0; i < 8; ++i) {
    if (i < n) {
      const float e = __expf(2.0f * x[i]);
      r[i] = 1.0f - 2.0f * __builtin_amdgcn_rcpf(e + 1.0f);
    }
  }
  return r;
}

static __device__ __forceinline__ v8f relu8(v8f x) {
  v8f r;
#pragma unroll
  for (int i = 0; i < 8; ++i) r[i] = fmaxf(x[i], 0.0f);
  return r;
}

// LN over features 0..3 (register-local thanks to transposed layout).
static __device__ __forceinline__ v8f layernorm4(v8f x, v8f w, v8f b) {
  const float mu = (x[0] + x[1] + x[2] + x[3]) * 0.25f;
  const float d0 = x[0] - mu, d1 = x[1] - mu, d2 = x[2] - mu, d3 = x[3] - mu;
  const float var = (d0 * d0 + d1 * d1 + d2 * d2 + d3 * d3) * 0.25f;
  const float rs = __builtin_amdgcn_rsqf(var + 1e-5f);
  v8f r = {};
  r[0] = d0 * rs * w[0] + b[0];
  r[1] = d1 * rs * w[1] + b[1];
  r[2] = d2 * rs * w[2] + b[2];
  r[3] = d3 * rs * w[3] + b[3];
  return r;  // features 4..7 forced to zero
}

__global__ void __launch_bounds__(256)
qcnn_fused_wmma(KParams p) {
  const int lane   = threadIdx.x & 31;
  const int wave   = (int)((blockIdx.x * blockDim.x + threadIdx.x) >> 5);
  const int nwaves = (int)((gridDim.x * blockDim.x) >> 5);
  const int ntiles = p.n_rows >> 4;

  // Hoisted weight A-operands (bias folded into augmented K column).
  const v16h A_fm = make_wA(p.fm_w, p.fm_b, 16,  8, lane);
  const v16h A_c1 = make_wA(p.c1_w, p.c1_b, 16, 16, lane);
  const v16h A_p1 = make_wA(p.p1_w, p.p1_b, 12, 16, lane);
  const v16h A_c2 = make_wA(p.c2_w, p.c2_b,  8, 12, lane);
  const v16h A_p2 = make_wA(p.p2_w, p.p2_b,  4,  8, lane);
  const v16h A_c3 = make_wA(p.c3_w, p.c3_b,  4,  4, lane);
  const v16h A_v  = make_wA(p.v_w,  p.v_b,   4,  4, lane);
  const v16h A_o  = make_wA(p.o_w,  p.o_b,   4,  4, lane);
  const v16h A_f1 = make_wA(p.f1_w, p.f1_b,  8,  4, lane);
  const v16h A_f2 = make_wA(p.f2_w, p.f2_b,  4,  8, lane);
  const v16h A_fs = make_wA(p.fs_w, p.fs_b,  4,  4, lane);
  const v16h A_hd = make_wA(p.hd_w, p.hd_b,  1,  4, lane);

  const v8f lnw1 = make_lnvec(p.ln1_w, 4, lane);
  const v8f lnb1 = make_lnvec(p.ln1_b, 4, lane);
  const v8f lnw2 = make_lnvec(p.ln2_w, 4, lane);
  const v8f lnb2 = make_lnvec(p.ln2_b, 4, lane);

  for (int tile = wave; tile < ntiles; tile += nwaves) {
    const int row = (tile << 4) + lane;  // valid for lanes 0..15

    // Prefetch next tile's input rows (global_prefetch_b8).
    const int nrow = row + (nwaves << 4);
    if (lane < 16 && nrow < p.n_rows)
      __builtin_prefetch(p.in + (size_t)nrow * 8, 0, 1);

    // B0 = X^T (8 features, augmented 1 at K=8), lanes 0..15 = batch rows.
    v16h xB = {};
    if (lane < 16) {
      const float4* s4 = (const float4*)(p.in + (size_t)row * 8);
      const float4 u = s4[0], w = s4[1];
      xB[0] = (_Float16)u.x; xB[1] = (_Float16)u.y;
      xB[2] = (_Float16)u.z; xB[3] = (_Float16)u.w;
      xB[4] = (_Float16)w.x; xB[5] = (_Float16)w.y;
      xB[6] = (_Float16)w.z; xB[7] = (_Float16)w.w;
    }
    xB[8] = (_Float16)1.0f;  // bias input (hi-lane slot = K24, don't-care)

    // Feature chain (6 tanh layers).
    v8f d = tanhN(wmma16(A_fm, xB), 8);            // 8 -> 16
    xB = pack_act(d, 16, lane);
    d = tanhN(wmma16(A_c1, xB), 8);                // 16 -> 16
    xB = pack_act(d, 16, lane);
    d = tanhN(wmma16(A_p1, xB), 8);                // 16 -> 12
    xB = pack_act(d, 12, lane);
    d = tanhN(wmma16(A_c2, xB), 8);                // 12 -> 8
    xB = pack_act(d, 8, lane);
    d = tanhN(wmma16(A_p2, xB), 4);                // 8 -> 4
    xB = pack_act(d, 4, lane);
    const v8f xs = tanhN(wmma16(A_c3, xB), 4);     // 4 -> 4  (== xs)

    // Attention, seq_len==1: softmax==1 => out == V; attn_out = o(v(xs)).
    v16h t = pack_act(xs, 4, lane);
    const v8f vv = wmma16(A_v, t);                 // V projection
    t = pack_act(vv, 4, lane);
    const v8f attn = wmma16(A_o, t);               // O projection

    const v8f h1 = layernorm4(xs + attn, lnw1, lnb1);

    // FFN 4 -> 8 -> 4.
    t = pack_act(h1, 4, lane);
    const v8f f1 = relu8(wmma16(A_f1, t));
    t = pack_act(f1, 8, lane);
    const v8f f2 = wmma16(A_f2, t);

    const v8f h2 = layernorm4(h1 + f2, lnw2, lnb2);

    // Final scale layer + sigmoid head.
    t = pack_act(h2, 4, lane);
    v8f fs = tanhN(wmma16(A_fs, t), 4);
#pragma unroll
    for (int i = 0; i < 4; ++i) fs[i] = fs[i] * 1.5f + 0.2f;
    t = pack_act(fs, 4, lane);
    const v8f hd = wmma16(A_hd, t);                // row 0 = logit per batch row

    const float logit = hd[0];
    const float y = __builtin_amdgcn_rcpf(1.0f + __expf(-logit));
    if (lane < 16 && row < p.n_rows) p.out[row] = y;
  }
}

extern "C" void kernel_launch(void* const* d_in, const int* in_sizes, int n_in,
                              void* d_out, int out_size, void* d_ws, size_t ws_size,
                              hipStream_t stream) {
  (void)n_in; (void)out_size; (void)d_ws; (void)ws_size;
  KParams p;
  p.in    = (const float*)d_in[0];
  p.fm_w  = (const float*)d_in[1];  p.fm_b  = (const float*)d_in[2];
  p.c1_w  = (const float*)d_in[3];  p.c1_b  = (const float*)d_in[4];
  p.p1_w  = (const float*)d_in[5];  p.p1_b  = (const float*)d_in[6];
  p.c2_w  = (const float*)d_in[7];  p.c2_b  = (const float*)d_in[8];
  p.p2_w  = (const float*)d_in[9];  p.p2_b  = (const float*)d_in[10];
  p.c3_w  = (const float*)d_in[11]; p.c3_b  = (const float*)d_in[12];
  // d_in[13..16] = q_w, q_b, k_w, k_b: mathematically dead (softmax over 1 elem).
  p.v_w   = (const float*)d_in[17]; p.v_b   = (const float*)d_in[18];
  p.o_w   = (const float*)d_in[19]; p.o_b   = (const float*)d_in[20];
  p.ln1_w = (const float*)d_in[21]; p.ln1_b = (const float*)d_in[22];
  p.ln2_w = (const float*)d_in[23]; p.ln2_b = (const float*)d_in[24];
  p.f1_w  = (const float*)d_in[25]; p.f1_b  = (const float*)d_in[26];
  p.f2_w  = (const float*)d_in[27]; p.f2_b  = (const float*)d_in[28];
  p.fs_w  = (const float*)d_in[29]; p.fs_b  = (const float*)d_in[30];
  p.hd_w  = (const float*)d_in[31]; p.hd_b  = (const float*)d_in[32];
  p.out   = (float*)d_out;
  p.n_rows = in_sizes[0] / 8;

  dim3 grid(1024), block(256);  // 8192 waves, ~16 row-tiles each (grid-stride)
  hipLaunchKernelGGL(qcnn_fused_wmma, grid, block, 0, stream, p);
}
